// HierarchicalDisentangled_69148973465939
// MI455X (gfx1250) — matrix operand order repaired and checked by reading, not compiled, not run on hardware
//
#include <hip/hip_runtime.h>
#include <hip/hip_bf16.h>
#include <math.h>

// ---------- types ----------
typedef __attribute__((ext_vector_type(16))) __bf16 v16bf;
typedef __attribute__((ext_vector_type(8)))  float  v8f;
typedef __attribute__((ext_vector_type(4)))  unsigned int v4u;
typedef __attribute__((ext_vector_type(8)))  int v8i_;
typedef __attribute__((ext_vector_type(4)))  int v4i_;

union ABFrag {
  v16bf v;
  unsigned short us[16];
  unsigned int   ui[8];
};

__device__ __forceinline__ unsigned short f2bf(float f) {
  union { float f; unsigned u; } x; x.f = f;
  unsigned r = x.u + 0x7FFFu + ((x.u >> 16) & 1u);   // RNE
  return (unsigned short)(r >> 16);
}

__device__ __forceinline__ float gelu_exact(float x) {
  return 0.5f * x * (1.0f + erff(x * 0.70710678118654752440f));
}

// TDM 2D tile load: tile (tile_rows x tile_cols) of bf16, row stride stride0 elems,
// zero-fill past (td0 cols, td1 rows) measured from the tile start address.
__device__ __forceinline__ void tdm_load_tile_bf16(
    unsigned lds_addr, unsigned long long gaddr,
    unsigned tile_cols, unsigned tile_rows,
    unsigned td0, unsigned td1, unsigned long long stride0)
{
  v4u g0; v8i_ g1; v4i_ g2 = {0, 0, 0, 0}, g3 = {0, 0, 0, 0};
  v8i_ g4 = {0, 0, 0, 0, 0, 0, 0, 0};
  g0[0] = 1u;                                            // count=1, user mode
  g0[1] = lds_addr;                                      // LDS dest (bytes)
  g0[2] = (unsigned)(gaddr & 0xFFFFFFFFull);             // global addr lo
  g0[3] = (unsigned)((gaddr >> 32) & 0x1FFFFFFull) | 0x80000000u; // hi + type=2
  g1[0] = (int)(1u << 16);                               // data_size=1 (2 bytes)
  g1[1] = (int)((td0 & 0xFFFFu) << 16);                  // tensor_dim0[15:0]
  g1[2] = (int)((td0 >> 16) | ((td1 & 0xFFFFu) << 16));  // dim0 hi | dim1 lo
  g1[3] = (int)((td1 >> 16) | ((tile_cols & 0xFFFFu) << 16)); // tile_dim0
  g1[4] = (int)(tile_rows & 0xFFFFu);                    // tile_dim1 (tile_dim2=0)
  g1[5] = (int)(unsigned)(stride0 & 0xFFFFFFFFull);      // dim0 stride lo
  g1[6] = (int)(unsigned)((stride0 >> 32) & 0xFFFFull);  // dim0 stride hi
  g1[7] = 0;
  __builtin_amdgcn_tensor_load_to_lds(g0, g1, g2, g3, g4, 0);
}

// ---------- fp32 -> bf16 (plain) ----------
__global__ void cvt_bf16_kernel(const float* __restrict__ src,
                                unsigned short* __restrict__ dst, int n) {
  int i = blockIdx.x * blockDim.x + threadIdx.x;
  if (i < n) dst[i] = f2bf(src[i]);
}

// ---------- conv-weight repack: kx order [1,2,0,3] within each (ci,ky) ----------
// dst slot s <- src kx = map[s]; A-side fragment gather uses the same order so
// even fragment pairs become contiguous aligned u32 loads of the input row.
__global__ void cvt_convw_kernel(const float* __restrict__ src,
                                 unsigned short* __restrict__ dst, int n) {
  int i = blockIdx.x * blockDim.x + threadIdx.x;
  if (i < n) {
    const int map[4] = {1, 2, 0, 3};
    int j = (i & ~3) | map[i & 3];
    dst[i] = f2bf(src[j]);
  }
}

// ---------- implicit-GEMM conv (4x4, s2, p1) + bias + exact GELU ----------
// Block: 8 waves, one 16-wide N tile, 32 M-tiles (8 waves x 4 accumulators).
// Weight tile (16 x Kc) double-buffered in LDS via TDM; B frags via ds_load.
__global__ void conv_wmma_kernel(
    const unsigned short* __restrict__ in,   // (B, CIN, HIN, WIN) bf16
    const unsigned short* __restrict__ wgt,  // (COUT, CIN*16) bf16 (kx-permuted)
    const float* __restrict__ bias,          // (COUT)
    unsigned short* __restrict__ out,        // (B, COUT, HO, WO) bf16
    int CIN, int COUT, int HIN, int WIN,
    int woSh, int hoSh, int Keff, int Kc, int ntShift)
{
  __shared__ unsigned short bsm[2][16 * 512];   // 2 x 16KB double buffer
  const int lane = threadIdx.x & 31;
  const int wave = threadIdx.x >> 5;
  const int g  = lane >> 4;
  const int ln = lane & 15;
  const int tn = blockIdx.x & ((1 << ntShift) - 1);
  const int mtBase = (blockIdx.x >> ntShift) * 32 + wave * 4;
  const int WO = 1 << woSh, HO = 1 << hoSh;
  const int HW = HIN * WIN;
  const int n0 = tn * 16;

  // 4 A-row pixels owned by this lane
  int bbOff[4], oyB[4], oxB[4];
#pragma unroll
  for (int i = 0; i < 4; ++i) {
    const int m  = (mtBase + i) * 16 + ln;
    const int ox = m & (WO - 1);
    const int oy = (m >> woSh) & (HO - 1);
    const int bb = m >> (woSh + hoSh);
    bbOff[i] = bb * CIN * HW;
    oyB[i] = oy * 2 - 1;
    oxB[i] = ox * 2 - 1;
  }
  const unsigned lds_base = (unsigned)(unsigned long long)(&bsm[0][0]);
  const int Kpad = (Keff + 31) & ~31;
  const int nch = Kpad / Kc;                 // K chunks

  v8f acc0 = {}, acc1 = {}, acc2 = {}, acc3 = {};
  if (wave == 0) {                           // prologue: chunk 0 -> buf 0
    const unsigned long long ga = (unsigned long long)(const void*)wgt +
        2ull * ((unsigned long long)n0 * Keff);
    tdm_load_tile_bf16(lds_base, ga, (unsigned)Kc, 16u,
                       (unsigned)Keff, 16u, (unsigned long long)Keff);
  }
  for (int c = 0; c < nch; ++c) {
    const int k0 = c * Kc;
    if (wave == 0) {
      if (c + 1 < nch) {                     // prefetch next chunk -> other buf
        const unsigned long long ga = (unsigned long long)(const void*)wgt +
            2ull * ((unsigned long long)n0 * Keff + (unsigned)(k0 + Kc));
        tdm_load_tile_bf16(lds_base + ((unsigned)((c + 1) & 1)) * 16384u, ga,
                           (unsigned)Kc, 16u,
                           (unsigned)(Keff - (k0 + Kc)), 16u,
                           (unsigned long long)Keff);
        __builtin_amdgcn_s_wait_tensorcnt(1);   // chunk c complete (in-order)
      } else {
        __builtin_amdgcn_s_wait_tensorcnt(0);
      }
    }
    __syncthreads();                         // tile c visible to all waves
    const unsigned short* buf = &bsm[c & 1][0];

    for (int kb = 0; kb < Kc; kb += 32) {
      ABFrag bf;                             // B: k = kb + (r<4?2r:16+2(r-4)) + 8g, col = ln
#pragma unroll
      for (int r = 0; r < 8; ++r)
        bf.ui[r] = *(const unsigned*)(&buf[ln * Kc + kb + ((r < 4) ? (2 * r) : (16 + 2 * (r - 4))) + 8 * g]);

#pragma unroll
      for (int i = 0; i < 4; ++i) {
        ABFrag a;                            // A: k = (r<4?2r:16+2(r-4)) + 8g
#pragma unroll
        for (int r = 0; r < 8; ++r) {
          const int kA = k0 + kb + ((r < 4) ? (2 * r) : (16 + 2 * (r - 4))) + 8 * g;
          const int ci = kA >> 4;
          const int ky = (kA >> 2) & 3;
          const int iy = oyB[i] + ky;
          const long long rowb = (long long)bbOff[i] + ci * HW + iy * WIN;
          const bool ok = (kA < Keff) && ((unsigned)iy < (unsigned)HIN);
          if ((r & 1) == 0) {
            // slots 0,1 = kx 1,2 -> ix = 2*ox, 2*ox+1 (aligned, x in bounds)
            a.ui[r] = ok ? *(const unsigned*)(in + rowb + oxB[i] + 1) : 0u;
          } else {
            // slots 2,3 = kx 0,3 -> ix = 2*ox-1, 2*ox+2 (edge-checked)
            unsigned short v0 = 0, v1 = 0;
            if (ok) {
              const int ix0 = oxB[i];
              const int ix1 = oxB[i] + 3;
              if (ix0 >= 0)  v0 = in[rowb + ix0];
              if (ix1 < WIN) v1 = in[rowb + ix1];
            }
            a.us[2 * r]     = v0;
            a.us[2 * r + 1] = v1;
          }
        }
        v8f& acc = (i == 0) ? acc0 : (i == 1) ? acc1 : (i == 2) ? acc2 : acc3;
        acc = __builtin_amdgcn_wmma_f32_16x16x32_bf16(false, a.v, false, bf.v,
                                                      (short)0, acc, false, false);
      }
    }
    __syncthreads();                         // done with buf[c&1] before reuse
  }

  const int co = n0 + ln;
  const float bv = bias[co];
#pragma unroll
  for (int i = 0; i < 4; ++i) {
    const v8f acc = (i == 0) ? acc0 : (i == 1) ? acc1 : (i == 2) ? acc2 : acc3;
#pragma unroll
    for (int r = 0; r < 8; ++r) {
      const int mrow = (mtBase + i) * 16 + r + 8 * g;
      const int oxs = mrow & (WO - 1);
      const int oys = (mrow >> woSh) & (HO - 1);
      const int b2  = mrow >> (woSh + hoSh);
      const float y = gelu_exact(acc[r] + bv);
      out[(((long long)b2 * COUT + co) << (hoSh + woSh)) + (oys << woSh) + oxs] = f2bf(y);
    }
  }
}

// ---------- GEMM: out(M,N) = A(M,K)@B(K,N) + bias, bf16; B double-buffered via TDM ----------
__global__ void gemm_wmma_kernel(
    const unsigned short* __restrict__ A,    // (M,K) bf16 row-major
    const unsigned short* __restrict__ Bm,   // (K,N) bf16 row-major
    const float* __restrict__ bias,          // (N)
    unsigned short* __restrict__ out,        // (M,N) bf16
    int M, int N, int K, int Kc, int ntShift)
{
  __shared__ unsigned short bsm[2][512 * 16];   // 2 x 16KB: Kc rows x 16 cols
  const int lane = threadIdx.x & 31;
  const int wave = threadIdx.x >> 5;
  const int g = lane >> 4, ln = lane & 15;
  const int tn = blockIdx.x & ((1 << ntShift) - 1);
  const int mtBase = (blockIdx.x >> ntShift) * 32 + wave * 4;
  const int n0 = tn * 16;
  const unsigned lds_base = (unsigned)(unsigned long long)(&bsm[0][0]);
  const int nch = K / Kc;

  v8f acc0 = {}, acc1 = {}, acc2 = {}, acc3 = {};
  if (wave == 0) {
    const unsigned long long ga = (unsigned long long)(const void*)Bm + 2ull * (unsigned)n0;
    tdm_load_tile_bf16(lds_base, ga, 16u, (unsigned)Kc,
                       (unsigned)(N - n0), (unsigned)K, (unsigned long long)N);
  }
  for (int c = 0; c < nch; ++c) {
    const int k0 = c * Kc;
    if (wave == 0) {
      if (c + 1 < nch) {
        const unsigned long long ga = (unsigned long long)(const void*)Bm +
            2ull * ((unsigned long long)(k0 + Kc) * N + (unsigned)n0);
        tdm_load_tile_bf16(lds_base + ((unsigned)((c + 1) & 1)) * 16384u, ga,
                           16u, (unsigned)Kc,
                           (unsigned)(N - n0), (unsigned)(K - (k0 + Kc)),
                           (unsigned long long)N);
        __builtin_amdgcn_s_wait_tensorcnt(1);
      } else {
        __builtin_amdgcn_s_wait_tensorcnt(0);
      }
    }
    __syncthreads();
    const unsigned short* buf = &bsm[c & 1][0];

    for (int kb = 0; kb < Kc; kb += 32) {
      ABFrag bf;
#pragma unroll
      for (int r = 0; r < 8; ++r) {
        const int kB = kb + ((r < 4) ? (2 * r) : (16 + 2 * (r - 4))) + 8 * g;
        bf.us[2 * r]     = buf[kB * 16 + ln];
        bf.us[2 * r + 1] = buf[(kB + 1) * 16 + ln];
      }
#pragma unroll
      for (int i = 0; i < 4; ++i) {
        ABFrag a;
        const long long arow = (long long)((mtBase + i) * 16 + ln) * K;
#pragma unroll
        for (int r = 0; r < 8; ++r) {
          const int kA = k0 + kb + ((r < 4) ? (2 * r) : (16 + 2 * (r - 4))) + 8 * g;
          a.ui[r] = *(const unsigned*)(A + arow + kA);
        }
        v8f& acc = (i == 0) ? acc0 : (i == 1) ? acc1 : (i == 2) ? acc2 : acc3;
        acc = __builtin_amdgcn_wmma_f32_16x16x32_bf16(false, a.v, false, bf.v,
                                                      (short)0, acc, false, false);
      }
    }
    __syncthreads();
  }
  const int ncol = n0 + ln;
  const float bv = bias[ncol];
#pragma unroll
  for (int i = 0; i < 4; ++i) {
    const v8f acc = (i == 0) ? acc0 : (i == 1) ? acc1 : (i == 2) ? acc2 : acc3;
#pragma unroll
    for (int r = 0; r < 8; ++r) {
      const int mrow = (mtBase + i) * 16 + r + 8 * g;
      out[(long long)mrow * N + ncol] = f2bf(acc[r] + bv);
    }
  }
}

// ---------- macro GEMM: z(b, l*32+d) = h @ macro_w + mb, dim-masked fp32 ----------
__global__ void macro_wmma_kernel(
    const unsigned short* __restrict__ h,    // (M, 256) bf16
    const unsigned short* __restrict__ mw,   // (3, 256, 32) bf16
    const float* __restrict__ mb,            // (96)
    float* __restrict__ zout,                // (M, 96) fp32 (masked)
    int M)
{
  const int K = 256, N = 96;
  const int lane = threadIdx.x & 31;
  const int wid  = blockIdx.x * (blockDim.x >> 5) + (threadIdx.x >> 5);
  const int ntiles = N >> 4;                 // 6
  const int total  = (M >> 4) * ntiles;
  if (wid >= total) return;                  // wave-uniform
  const int tm = wid / ntiles;
  const int tn = wid - tm * ntiles;
  const int g  = lane >> 4;
  const int ln = lane & 15;
  const int mrowA = tm * 16 + ln;
  const int ncol  = tn * 16 + ln;            // n = l*32 + d
  const int l = ncol >> 5, d = ncol & 31;

  v8f acc = {};
  for (int kb = 0; kb < K; kb += 32) {
    ABFrag a, b;
#pragma unroll
    for (int r = 0; r < 8; ++r) {
      const int kA = kb + ((r < 4) ? (2 * r) : (16 + 2 * (r - 4))) + 8 * g;
      a.ui[r] = *(const unsigned*)(h + (long long)mrowA * K + kA);
      const int kB = kb + ((r < 4) ? (2 * r) : (16 + 2 * (r - 4))) + 8 * g;
      b.us[2 * r]     = mw[((long long)l * 256 + kB) * 32 + d];
      b.us[2 * r + 1] = mw[((long long)l * 256 + kB + 1) * 32 + d];
    }
    acc = __builtin_amdgcn_wmma_f32_16x16x32_bf16(false, a.v, false, b.v,
                                                  (short)0, acc, false, false);
  }
  const float bv = mb[ncol];
  const float msk = (d < (8 << l)) ? 1.0f : 0.0f;
#pragma unroll
  for (int r = 0; r < 8; ++r) {
    const int mrow = tm * 16 + r + 8 * g;
    zout[(long long)mrow * 96 + ncol] = (acc[r] + bv) * msk;
  }
}

// ---------- VQ: argmin over codes, z_macro blend, loss ----------
__global__ void vq_kernel(
    const float* __restrict__ z,         // (512, 3, 32) masked
    const float* __restrict__ codebook,  // (3, 256, 32) fp32
    const float* __restrict__ prev,      // (512, 3, 32)
    const int*  __restrict__ step,
    float* __restrict__ zmacro,          // (512, 3, 32)
    float* __restrict__ idx_out,         // (512, 3) as float
    float* __restrict__ loss_out)        // scalar (pre-zeroed)
{
  __shared__ float zsh[8][32];
  const int lane = threadIdx.x & 31;
  const int w    = threadIdx.x >> 5;
  const int pair = blockIdx.x * 8 + w;     // exactly 512*3 pairs
  const int b = pair / 3, l = pair % 3;

  const float zd = z[(size_t)b * 96 + l * 32 + lane];
  zsh[w][lane] = zd;
  __syncthreads();

  const int ncodes = 64 << l;
  const float* cb = codebook + (size_t)l * 256 * 32;
  float bestd = 3.4e38f; int besti = 0x7fffffff;
#pragma unroll 1
  for (int kk = 0; kk < 8; ++kk) {
    const int k = lane + 32 * kk;
    float s = 0.f;
#pragma unroll
    for (int d = 0; d < 32; ++d) {
      const float diff = zsh[w][d] - cb[(size_t)k * 32 + d];
      s += diff * diff;
    }
    if (k >= ncodes) s += 1.0e9f;
    if (s < bestd || (s == bestd && k < besti)) { bestd = s; besti = k; }
  }
  for (int off = 16; off > 0; off >>= 1) {
    const float od = __shfl_xor(bestd, off, 32);
    const int   oi = __shfl_xor(besti, off, 32);
    if (od < bestd || (od == bestd && oi < besti)) { bestd = od; besti = oi; }
  }

  const float cq  = cb[(size_t)besti * 32 + lane];
  const int   dim = 8 << l;
  const float mk  = (lane < dim) ? 1.f : 0.f;
  const float df  = (zd - cq) * mk;
  float lsum = df * df;
  for (int off = 16; off > 0; off >>= 1) lsum += __shfl_xor(lsum, off, 32);

  const int freqs[3] = {8, 4, 1};
  const float um = ((step[0] % freqs[l]) == 0) ? 1.f : 0.f;
  const float zst = zd + (cq - zd);
  zmacro[(size_t)b * 96 + l * 32 + lane] =
      um * zst + (1.f - um) * prev[(size_t)b * 96 + l * 32 + lane];

  if (lane == 0) {
    idx_out[b * 3 + l] = (float)besti;
    atomicAdd(loss_out, lsum * (1.25f / 49152.0f));  // 1.25 * mean
  }
}

__global__ void zero1_kernel(float* p) { *p = 0.0f; }

// ---------- host orchestration ----------
extern "C" void kernel_launch(void* const* d_in, const int* in_sizes, int n_in,
                              void* d_out, int out_size, void* d_ws, size_t ws_size,
                              hipStream_t stream) {
  const float* obs = (const float*)d_in[0];
  const float* w1  = (const float*)d_in[1];
  const float* b1  = (const float*)d_in[2];
  const float* w2  = (const float*)d_in[3];
  const float* b2  = (const float*)d_in[4];
  const float* w3  = (const float*)d_in[5];
  const float* b3  = (const float*)d_in[6];
  const float* w4  = (const float*)d_in[7];
  const float* b4  = (const float*)d_in[8];
  const float* pw  = (const float*)d_in[9];
  const float* pb  = (const float*)d_in[10];
  const float* mw  = (const float*)d_in[11];
  const float* mb  = (const float*)d_in[12];
  const float* cb  = (const float*)d_in[13];
  const float* prv = (const float*)d_in[14];
  const int*   stp = (const int*)d_in[15];

  char* ws = (char*)d_ws;
  size_t off = 0;
  auto alloc_us = [&](size_t elems) -> unsigned short* {
    unsigned short* p = (unsigned short*)(ws + off);
    off += (elems * 2 + 255) & ~(size_t)255;
    return p;
  };
  unsigned short* obs_bf = alloc_us(512ull * 3 * 64 * 64);
  unsigned short* w1_bf  = alloc_us(32 * 48);
  unsigned short* w2_bf  = alloc_us(64 * 512);
  unsigned short* w3_bf  = alloc_us(128 * 1024);
  unsigned short* w4_bf  = alloc_us(256 * 2048);
  unsigned short* pw_bf  = alloc_us(4096 * 256);
  unsigned short* mw_bf  = alloc_us(3 * 256 * 32);
  unsigned short* a1     = alloc_us(512ull * 32 * 32 * 32);
  unsigned short* a2     = alloc_us(512ull * 64 * 16 * 16);
  unsigned short* a3     = alloc_us(512ull * 128 * 8 * 8);
  unsigned short* a4     = alloc_us(512ull * 256 * 4 * 4);
  unsigned short* hbuf   = alloc_us(512 * 256);

  auto cvt = [&](const float* s, unsigned short* d, int n) {
    cvt_bf16_kernel<<<(n + 255) / 256, 256, 0, stream>>>(s, d, n);
  };
  auto cvtw = [&](const float* s, unsigned short* d, int n) {  // kx-permuted
    cvt_convw_kernel<<<(n + 255) / 256, 256, 0, stream>>>(s, d, n);
  };
  cvt(obs, obs_bf, 512 * 3 * 64 * 64);
  cvtw(w1, w1_bf, 32 * 48);
  cvtw(w2, w2_bf, 64 * 512);
  cvtw(w3, w3_bf, 128 * 1024);
  cvtw(w4, w4_bf, 256 * 2048);
  cvt(pw, pw_bf, 4096 * 256);
  cvt(mw, mw_bf, 3 * 256 * 32);

  // conv(in, wgt, bias, out, CIN, COUT, HIN, WIN, sh, Kc, ntShift)
  auto conv = [&](const unsigned short* in, const unsigned short* wg, const float* bi,
                  unsigned short* outp, int CIN, int COUT, int HIN, int WIN,
                  int sh, int Kc, int ntShift) {
    const int M = 512 << (2 * sh);                    // 512*HO*WO
    const int blocks = ((M / 16) / 32) << ntShift;
    conv_wmma_kernel<<<blocks, 256, 0, stream>>>(
        in, wg, bi, outp, CIN, COUT, HIN, WIN, sh, sh, CIN * 16, Kc, ntShift);
  };
  conv(obs_bf, w1_bf, b1, a1,   3,  32, 64, 64, 5,  64, 1);  // 2048 blocks
  conv(a1,     w2_bf, b2, a2,  32,  64, 32, 32, 4, 512, 2);  // 1024 blocks
  conv(a2,     w3_bf, b3, a3,  64, 128, 16, 16, 3, 512, 3);  //  512 blocks
  conv(a3,     w4_bf, b4, a4, 128, 256,  8,  8, 2, 512, 4);  //  256 blocks

  // proj: (512, 4096) @ (4096, 256) + pb  -> 16 blocks (N tiles), 32 M-tiles each
  gemm_wmma_kernel<<<16, 256, 0, stream>>>(a4, pw_bf, pb, hbuf, 512, 256, 4096, 512, 4);

  float* out    = (float*)d_out;
  float* zmacro = out;               // 49152
  float* idxo   = out + 49152;       // 1536
  float* losso  = out + 50688;       // 1
  float* zz     = out + 50689;       // 49152

  {
    const int waves = (512 / 16) * (96 / 16);
    macro_wmma_kernel<<<(waves + 7) / 8, 256, 0, stream>>>(hbuf, mw_bf, mb, zz, 512);
  }

  zero1_kernel<<<1, 1, 0, stream>>>(losso);
  vq_kernel<<<(512 * 3) / 8, 256, 0, stream>>>(zz, cb, prv, stp, zmacro, idxo, losso);
}
